// MultiHeadSelfAttention_7447473291694
// MI455X (gfx1250) — matrix operand order repaired and checked by reading, not compiled, not run on hardware
//
#include <hip/hip_runtime.h>
#include <cstdint>
#include <cstddef>

// ---------------- types ----------------
typedef _Float16 v16h __attribute__((ext_vector_type(16)));
typedef _Float16 h8   __attribute__((ext_vector_type(8)));
typedef _Float16 h4   __attribute__((ext_vector_type(4)));
typedef float    v8f  __attribute__((ext_vector_type(8)));
typedef float    f4   __attribute__((ext_vector_type(4)));
typedef uint32_t u32x4 __attribute__((ext_vector_type(4)));

union Frag16 { v16h h; u32x4 u[2]; };

#define WMMA16(A, B, C) \
  __builtin_amdgcn_wmma_f32_16x16x32_f16(false, (A), false, (B), (short)0, (C), false, false)

// ---------------- problem constants ----------------
constexpr int kB  = 4;
constexpr int kN  = 2048;
constexpr int kD  = 512;
constexpr int kH  = 8;
constexpr int kHD = 64;
constexpr int kM  = kB * kN;     // 8192 rows

// ---------------- kernel 0: f32 -> f16 convert ----------------
__global__ void cvt_f32_f16(const float* __restrict__ in,
                            _Float16* __restrict__ out, int n4) {
  int i = blockIdx.x * blockDim.x + threadIdx.x;
  if (i < n4) {
    f4 v = *reinterpret_cast<const f4*>(in + (size_t)i * 4);
    h4 h = __builtin_convertvector(v, h4);
    *reinterpret_cast<h4*>(out + (size_t)i * 4) = h;
  }
}

// ---------------- generic WMMA GEMM: C = A(f16) * B(f32) + bias ----------------
// Block: 128 threads (4 waves). Tile: 128(M) x 64(N). Wave w: rows [32w, 32w+32).
// EPI==0: scatter f16 into Q/K/V head layout [B,H,N,HD].  EPI==1: f32 row-major out.
template <int EPI>
__global__ __launch_bounds__(128) void gemm16(
    const _Float16* __restrict__ A, const float* __restrict__ Bw,
    const float* __restrict__ bias,
    _Float16* __restrict__ q, _Float16* __restrict__ k, _Float16* __restrict__ v,
    float* __restrict__ O, int Kd, int Nw) {
  __shared__ __align__(16) _Float16 Bt[64][40];  // transposed weight tile [n][k]

  const int tid = threadIdx.x;
  const int w = tid >> 5, l = tid & 31;
  const int lo = l & 15, hi = l >> 4;
  const int m0 = blockIdx.y * 128, n0 = blockIdx.x * 64;

  v8f acc[2][4];
#pragma unroll
  for (int rg = 0; rg < 2; rg++)
#pragma unroll
    for (int nt = 0; nt < 4; nt++)
#pragma unroll
      for (int r = 0; r < 8; r++) acc[rg][nt][r] = 0.0f;

  const _Float16* Arow0 = A + (size_t)(m0 + w * 32 + lo) * Kd;
  const _Float16* Arow1 = Arow0 + (size_t)16 * Kd;

  for (int k0 = 0; k0 < Kd; k0 += 32) {
    // stage 32(k) x 64(n) f32 weights -> LDS transposed f16
#pragma unroll
    for (int i = 0; i < 4; i++) {
      int idx = tid + 128 * i;          // 0..511
      int kk = idx >> 4;                // 0..31
      int n4 = (idx & 15) * 4;          // 0..60
      f4 val = *reinterpret_cast<const f4*>(Bw + (size_t)(k0 + kk) * Nw + n0 + n4);
      h4 hv = __builtin_convertvector(val, h4);
      Bt[n4 + 0][kk] = hv.x;
      Bt[n4 + 1][kk] = hv.y;
      Bt[n4 + 2][kk] = hv.z;
      Bt[n4 + 3][kk] = hv.w;
    }
    __syncthreads();

    // batch-load all fragments for this K-step, then burst the WMMAs
    Frag16 a[2];
    a[0].u[0] = *reinterpret_cast<const u32x4*>(Arow0 + k0 + hi * 8);
    a[0].u[1] = *reinterpret_cast<const u32x4*>(Arow0 + k0 + 16 + hi * 8);
    a[1].u[0] = *reinterpret_cast<const u32x4*>(Arow1 + k0 + hi * 8);
    a[1].u[1] = *reinterpret_cast<const u32x4*>(Arow1 + k0 + 16 + hi * 8);

    Frag16 b[4];
#pragma unroll
    for (int nt = 0; nt < 4; nt++) {
      const _Float16* bp = &Bt[nt * 16 + lo][hi * 8];
      b[nt].u[0] = *reinterpret_cast<const u32x4*>(bp);
      b[nt].u[1] = *reinterpret_cast<const u32x4*>(bp + 16);
    }

#pragma unroll
    for (int rg = 0; rg < 2; rg++)
#pragma unroll
      for (int nt = 0; nt < 4; nt++)
        acc[rg][nt] = WMMA16(a[rg].h, b[nt].h, acc[rg][nt]);
    __syncthreads();
  }

  // epilogue
#pragma unroll
  for (int rg = 0; rg < 2; rg++) {
#pragma unroll
    for (int nt = 0; nt < 4; nt++) {
      int col = n0 + nt * 16 + lo;
      float bv = bias[col];
#pragma unroll
      for (int r = 0; r < 8; r++) {
        int row = m0 + w * 32 + rg * 16 + r + hi * 8;
        float val = acc[rg][nt][r] + bv;
        if (EPI == 0) {
          int part = col >> 9;         // 0=q 1=k 2=v
          int cc = col & 511;
          int hh = cc >> 6, d = cc & 63;
          int bb = row >> 11, nn = row & 2047;
          size_t dst = ((size_t)(bb * kH + hh) * kN + nn) * kHD + d;
          _Float16 hv = (_Float16)val;
          if (part == 0) q[dst] = hv;
          else if (part == 1) k[dst] = hv;
          else v[dst] = hv;
        } else {
          O[(size_t)row * Nw + col] = val;
        }
      }
    }
  }
}

// ---------------- flash attention: per (b,h), 64 query rows / block ----------------
__global__ __launch_bounds__(128) void attn64(
    const _Float16* __restrict__ Qh, const _Float16* __restrict__ Kh,
    const _Float16* __restrict__ Vh, _Float16* __restrict__ Ctx) {
  __shared__ __align__(16) _Float16 Vt[64][72];      // V tile transposed [hd][key]
  __shared__ __align__(16) _Float16 Pt[4][16][72];   // per-wave exp(P) tile

  const int tid = threadIdx.x;
  const int w = tid >> 5, l = tid & 31;
  const int lo = l & 15, hi = l >> 4;
  const int bh = blockIdx.y;            // b*H + h
  const int m0 = blockIdx.x * 64;
  const size_t base = (size_t)bh * kN * kHD;
  const _Float16* Q = Qh + base;
  const _Float16* K = Kh + base;
  const _Float16* V = Vh + base;

  // preload Q fragments (head-dim 0..31 and 32..63) for this wave's 16 rows
  Frag16 qa0, qa1;
  {
    const _Float16* qr = Q + (size_t)(m0 + w * 16 + lo) * kHD;
    qa0.u[0] = *reinterpret_cast<const u32x4*>(qr + hi * 8);
    qa0.u[1] = *reinterpret_cast<const u32x4*>(qr + 16 + hi * 8);
    qa1.u[0] = *reinterpret_cast<const u32x4*>(qr + 32 + hi * 8);
    qa1.u[1] = *reinterpret_cast<const u32x4*>(qr + 48 + hi * 8);
  }

  v8f acc[4];
  float rowm[8], rowl[8];
#pragma unroll
  for (int hdt = 0; hdt < 4; hdt++)
#pragma unroll
    for (int r = 0; r < 8; r++) acc[hdt][r] = 0.0f;
#pragma unroll
  for (int r = 0; r < 8; r++) { rowm[r] = -3.0e38f; rowl[r] = 0.0f; }

  for (int j = 0; j < kN / 64; j++) {
    // prefetch next K tile into cache hierarchy (global_prefetch_b8)
    if (j + 1 < kN / 64)
      __builtin_prefetch(K + (size_t)((j + 1) * 64 + l) * kHD, 0, 1);

    // stage V tile (64 keys x 64 hd) transposed into LDS
#pragma unroll
    for (int i = 0; i < 4; i++) {
      int idx = tid + 128 * i;          // 0..511 octets
      int key = idx >> 3;               // 0..63
      int hdo = (idx & 7) * 8;          // 0..56
      h8 vv = *reinterpret_cast<const h8*>(V + (size_t)(j * 64 + key) * kHD + hdo);
#pragma unroll
      for (int t = 0; t < 8; t++) Vt[hdo + t][key] = vv[t];
    }
    __syncthreads();

    // --- S = (Q K^T)/8 : batch all 16 K-fragment loads, then 8 WMMAs ---
    Frag16 kb[4][2];
#pragma unroll
    for (int nt = 0; nt < 4; nt++) {
      const _Float16* kr = K + (size_t)(j * 64 + nt * 16 + lo) * kHD;
      kb[nt][0].u[0] = *reinterpret_cast<const u32x4*>(kr + hi * 8);
      kb[nt][0].u[1] = *reinterpret_cast<const u32x4*>(kr + 16 + hi * 8);
      kb[nt][1].u[0] = *reinterpret_cast<const u32x4*>(kr + 32 + hi * 8);
      kb[nt][1].u[1] = *reinterpret_cast<const u32x4*>(kr + 48 + hi * 8);
    }
    v8f s[4];
#pragma unroll
    for (int nt = 0; nt < 4; nt++) {
      v8f z;
#pragma unroll
      for (int r = 0; r < 8; r++) z[r] = 0.0f;
      z = WMMA16(qa0.h, kb[nt][0].h, z);
      z = WMMA16(qa1.h, kb[nt][1].h, z);
#pragma unroll
      for (int r = 0; r < 8; r++) s[nt][r] = z[r] * 0.125f;
    }

    // --- online softmax (row stats shared across each 16-lane half) ---
    float tmax[8];
#pragma unroll
    for (int r = 0; r < 8; r++)
      tmax[r] = fmaxf(fmaxf(s[0][r], s[1][r]), fmaxf(s[2][r], s[3][r]));
#pragma unroll
    for (int off = 1; off < 16; off <<= 1)
#pragma unroll
      for (int r = 0; r < 8; r++)
        tmax[r] = fmaxf(tmax[r], __shfl_xor(tmax[r], off, 32));

    float alpha[8];
#pragma unroll
    for (int r = 0; r < 8; r++) {
      float mn = fmaxf(rowm[r], tmax[r]);
      alpha[r] = __expf(rowm[r] - mn);
      rowm[r] = mn;
    }
#pragma unroll
    for (int hdt = 0; hdt < 4; hdt++)
#pragma unroll
      for (int r = 0; r < 8; r++) acc[hdt][r] *= alpha[r];

    float psum[8];
#pragma unroll
    for (int r = 0; r < 8; r++) psum[r] = 0.0f;
#pragma unroll
    for (int nt = 0; nt < 4; nt++) {
#pragma unroll
      for (int r = 0; r < 8; r++) {
        float p = __expf(s[nt][r] - rowm[r]);
        psum[r] += p;
        Pt[w][r + hi * 8][nt * 16 + lo] = (_Float16)p;
      }
    }
#pragma unroll
    for (int off = 1; off < 16; off <<= 1)
#pragma unroll
      for (int r = 0; r < 8; r++) psum[r] += __shfl_xor(psum[r], off, 32);
#pragma unroll
    for (int r = 0; r < 8; r++) rowl[r] = rowl[r] * alpha[r] + psum[r];

    // --- context += P @ V : batch all 20 LDS fragment loads, then 8 WMMAs ---
    Frag16 pa[2];
#pragma unroll
    for (int kc = 0; kc < 2; kc++) {
      const _Float16* pr = &Pt[w][lo][kc * 32 + hi * 8];
      pa[kc].u[0] = *reinterpret_cast<const u32x4*>(pr);
      pa[kc].u[1] = *reinterpret_cast<const u32x4*>(pr + 16);
    }
    Frag16 vb[2][4];
#pragma unroll
    for (int kc = 0; kc < 2; kc++)
#pragma unroll
      for (int hdt = 0; hdt < 4; hdt++) {
        const _Float16* vp = &Vt[hdt * 16 + lo][kc * 32 + hi * 8];
        vb[kc][hdt].u[0] = *reinterpret_cast<const u32x4*>(vp);
        vb[kc][hdt].u[1] = *reinterpret_cast<const u32x4*>(vp + 16);
      }
#pragma unroll
    for (int kc = 0; kc < 2; kc++)
#pragma unroll
      for (int hdt = 0; hdt < 4; hdt++)
        acc[hdt] = WMMA16(pa[kc].h, vb[kc][hdt].h, acc[hdt]);
    __syncthreads();
  }

  // normalize + write context in [B,N,H*HD] layout (f16) for the proj GEMM
  const int b = bh >> 3, h = bh & 7;
#pragma unroll
  for (int r = 0; r < 8; r++) {
    float inv = 1.0f / rowl[r];
    int row = m0 + w * 16 + r + hi * 8;
    _Float16* dst = Ctx + ((size_t)(b * kN + row)) * kD + h * kHD;
#pragma unroll
    for (int hdt = 0; hdt < 4; hdt++)
      dst[hdt * 16 + lo] = (_Float16)(acc[hdt][r] * inv);
  }
}

// ---------------- host launcher ----------------
extern "C" void kernel_launch(void* const* d_in, const int* in_sizes, int n_in,
                              void* d_out, int out_size, void* d_ws, size_t ws_size,
                              hipStream_t stream) {
  const float* x      = (const float*)d_in[0];  // [B,N,D]
  const float* w_qkv  = (const float*)d_in[1];  // [D,3D]
  const float* b_qkv  = (const float*)d_in[2];  // [3D]
  const float* w_proj = (const float*)d_in[3];  // [D,D]
  const float* b_proj = (const float*)d_in[4];  // [D]
  float* out = (float*)d_out;                   // [B,N,D]

  const size_t chunk = (size_t)kM * kD;         // 4,194,304 elements
  _Float16* Xh  = (_Float16*)d_ws;
  _Float16* Qh  = Xh + chunk;
  _Float16* Kh  = Qh + chunk;
  _Float16* Vh  = Kh + chunk;
  _Float16* Ctx = Vh + chunk;                   // total 40 MB of workspace

  // 1) x -> f16
  int n4 = (int)(chunk / 4);
  cvt_f32_f16<<<(n4 + 255) / 256, 256, 0, stream>>>(x, Xh, n4);

  // 2) QKV projection GEMM: [8192,512] x [512,1536] -> Q/K/V head layout f16
  gemm16<0><<<dim3((3 * kD) / 64, kM / 128), 128, 0, stream>>>(
      Xh, w_qkv, b_qkv, Qh, Kh, Vh, nullptr, kD, 3 * kD);

  // 3) flash attention per (b,h): grid = (row blocks, B*H)
  attn64<<<dim3(kN / 64, kB * kH), 128, 0, stream>>>(Qh, Kh, Vh, Ctx);

  // 4) output projection GEMM: [8192,512] x [512,512] + bias -> f32 out
  gemm16<1><<<dim3(kD / 64, kM / 128), 128, 0, stream>>>(
      Ctx, w_proj, b_proj, nullptr, nullptr, nullptr, out, kD, kD);
}
